// CrossNetMoE_83657372992063
// MI455X (gfx1250) — matrix operand order repaired and checked by reading, not compiled, not run on hardware
//
#include <hip/hip_runtime.h>
#include <hip/hip_bf16.h>
#include <math.h>

// ---- problem constants (match reference) ----
#define NB 16384   // batch
#define ND 1024    // embed dim
#define NR 64      // low rank
#define NE 4       // experts
#define NL 3       // layers
#define TB1 64     // batch rows per workgroup (kernel 1)
#define TB2 128    // batch rows per workgroup (kernel 2)
#define KC 32      // K-chunk for bf16 WMMA (16x16x32)

#if __has_builtin(__builtin_amdgcn_global_load_async_to_lds_b128)
#define ASYNC_COPY 1
#pragma message("CDNA5: using global_load_async_to_lds_b128 staging path")
#else
#define ASYNC_COPY 0
#pragma message("CDNA5: async-to-LDS builtin unavailable; VGPR staging fallback")
#endif

typedef __attribute__((ext_vector_type(16))) __bf16 v16bf;
typedef __attribute__((ext_vector_type(8)))  float  v8f;
typedef int v4i_ __attribute__((vector_size(16)));   // matches builtin param type
typedef unsigned short bf16_t;

union Frag { unsigned u[8]; v16bf h; };

__device__ __forceinline__ unsigned short f2bf(float f) {
  union { float f; unsigned u; } v; v.f = f;
  unsigned r = v.u + 0x7FFFu + ((v.u >> 16) & 1u);   // round-to-nearest-even
  return (unsigned short)(r >> 16);
}
__device__ __forceinline__ unsigned pk2(float lo, float hi) {
  return (unsigned)f2bf(lo) | ((unsigned)f2bf(hi) << 16);
}
__device__ __forceinline__ v8f vzero() {
  v8f z;
#pragma unroll
  for (int i = 0; i < 8; ++i) z[i] = 0.f;
  return z;
}

#if ASYNC_COPY
#define ASYNC_B128(G, L)                                          \
  __builtin_amdgcn_global_load_async_to_lds_b128(                 \
      (__attribute__((address_space(1))) v4i_*)(G),               \
      (__attribute__((address_space(3))) v4i_*)(L), 0, 0)
__device__ __forceinline__ void wait_async0() {
#if __has_builtin(__builtin_amdgcn_s_wait_asynccnt)
  __builtin_amdgcn_s_wait_asynccnt(0);
#else
  asm volatile("s_wait_asynccnt 0x0" ::: "memory");
#endif
}
#endif

// ============================================================================
// Prologue: f32 -> bf16 conversion (x, V, U, C converted once per launch).
// ============================================================================
__global__ __launch_bounds__(256) void cvt_f32_bf16_kernel(
    const float* __restrict__ src, bf16_t* __restrict__ dst, int n) {
  int i = (blockIdx.x * 256 + threadIdx.x) * 4;
  if (i < n) {
    float4 v = *(const float4*)(src + i);
    unsigned* d = (unsigned*)(dst + i);
    d[0] = pk2(v.x, v.y);
    d[1] = pk2(v.z, v.w);
  }
}

// ============================================================================
// Kernel 1: c = tanh(C · tanh(V · xl)) (bf16 out), g = softmax(Wg · xl).
// One WG = 64 batch rows, 8 waves.  Wave w: row-block (w&3), expert pair (w>>2)
// => 8 f32 accumulators per wave (64 VGPRs).
// Double-buffered async-DMA LDS staging.
// ============================================================================
__global__ __launch_bounds__(256)
__attribute__((amdgpu_waves_per_eu(1)))
void crossnet_vcg_kernel(
    const float*  __restrict__ xin,       // [B,D] f32 (gating path)
    const bf16_t* __restrict__ xbf,       // [B,D] bf16 (GEMM A operand)
    const bf16_t* __restrict__ Vbf,       // [E,R,D] bf16
    const bf16_t* __restrict__ Cbf,       // [E,R,R] bf16
    const float*  __restrict__ Wg,        // [E,D] f32
    bf16_t* __restrict__ c_out,           // [B,E,R] bf16
    float*  __restrict__ g_out)           // [B,E] f32
{
  __shared__ bf16_t lds_x[2][TB1 * KC];        // 2 x 4 KB
  __shared__ bf16_t lds_v[2][NE * NR * KC];    // 2 x 16 KB
  __shared__ bf16_t lds_s[2][TB1 * NR];        // per expert-pair v staging, 16 KB

  const int tid  = threadIdx.x;
  const int lane = tid & 31;
  const int wave = tid >> 5;
  const int rb   = wave & 3;     // row block (16 rows)
  const int ep   = wave >> 2;    // expert pair: experts ep*2, ep*2+1
  const int row0 = blockIdx.x * TB1;

  // ---------- gating: threads 0..63, one row each (f32, L2-hot) ----------
  if (tid < TB1) {
    const float* xr = xin + (size_t)(row0 + tid) * ND;
    float s0 = 0.f, s1 = 0.f, s2 = 0.f, s3 = 0.f;
    for (int d = 0; d < ND; d += 4) {
      float4 xv = *(const float4*)(xr + d);
      float4 w0 = *(const float4*)(Wg + 0 * ND + d);
      float4 w1 = *(const float4*)(Wg + 1 * ND + d);
      float4 w2 = *(const float4*)(Wg + 2 * ND + d);
      float4 w3 = *(const float4*)(Wg + 3 * ND + d);
      s0 += xv.x * w0.x + xv.y * w0.y + xv.z * w0.z + xv.w * w0.w;
      s1 += xv.x * w1.x + xv.y * w1.y + xv.z * w1.z + xv.w * w1.w;
      s2 += xv.x * w2.x + xv.y * w2.y + xv.z * w2.z + xv.w * w2.w;
      s3 += xv.x * w3.x + xv.y * w3.y + xv.z * w3.z + xv.w * w3.w;
    }
    float m = fmaxf(fmaxf(s0, s1), fmaxf(s2, s3));
    float e0 = __expf(s0 - m), e1 = __expf(s1 - m);
    float e2 = __expf(s2 - m), e3 = __expf(s3 - m);
    float inv = 1.f / (e0 + e1 + e2 + e3);
    float* gp = g_out + (size_t)(row0 + tid) * NE;
    gp[0] = e0 * inv; gp[1] = e1 * inv; gp[2] = e2 * inv; gp[3] = e3 * inv;
  }

  // ---------- v = tanh(x @ V^T), WMMA over K = D in chunks of 32 ----------
  v8f acc[2][4];
#pragma unroll
  for (int ee = 0; ee < 2; ++ee)
#pragma unroll
    for (int nt = 0; nt < 4; ++nt) acc[ee][nt] = vzero();

  // A-frag (16-bit A 16x32): M = lane&15, K = (lane>=16?8:0)+(j>=4?16:0)+(j&3)*2
  const int mrow = rb * 16 + (lane & 15);
  const int kbA  = (lane >> 4) * 8;
  // B-frag (16-bit B 32x16): N = lane&15, K = (lane>=16?8:0)+(j>=4?16:0)+(j&3)*2
  const int nB  = lane & 15;
  const int kbB = (lane >> 4) * 8;

  // staging: x -> 4 threads per row, 8 bf16 each; V -> one (e,n) row per thread
  const int sxr = tid >> 2;
  const int sxk = (tid & 3) * 8;
  const bf16_t* gx = xbf + (size_t)(row0 + sxr) * ND + sxk;  // + k0
  const bf16_t* gv = Vbf + (size_t)tid * ND;                 // + k0

#if ASYNC_COPY
  auto stage = [&](int buf, int k0) {
    ASYNC_B128(gx + k0, &lds_x[buf][sxr * KC + sxk]);
    const bf16_t* g1 = gv + k0;
    bf16_t* l1 = &lds_v[buf][tid * KC];
#pragma unroll
    for (int q = 0; q < 4; ++q) ASYNC_B128(g1 + q * 8, l1 + q * 8);
  };
  stage(0, 0);
#endif

  for (int k0 = 0, it = 0; k0 < ND; k0 += KC, ++it) {
    const int buf = it & 1;
#if ASYNC_COPY
    wait_async0();        // my async copies into `buf` have landed
    __syncthreads();      // => everyone's landed; prior reads of buf^1 done
    if (k0 + KC < ND) stage(buf ^ 1, k0 + KC);  // DMA next chunk during WMMAs
#else
    { // VGPR staging fallback (pure bf16 copy)
      const uint4* s0 = (const uint4*)(gx + k0);
      *(uint4*)&lds_x[buf][sxr * KC + sxk] = s0[0];
      const uint4* s1 = (const uint4*)(gv + k0);
      uint4* d1 = (uint4*)&lds_v[buf][tid * KC];
      d1[0] = s1[0]; d1[1] = s1[1]; d1[2] = s1[2]; d1[3] = s1[3];
    }
    __syncthreads();
#endif

    const unsigned* lx32 = (const unsigned*)lds_x[buf];
    const unsigned* lv32 = (const unsigned*)lds_v[buf];
    Frag a;
#pragma unroll
    for (int j = 0; j < 8; ++j) {
      int k = kbA + ((j & 4) ? 16 : 0) + (j & 3) * 2;
      a.u[j] = lx32[(mrow * KC + k) >> 1];
    }
#pragma unroll
    for (int ee = 0; ee < 2; ++ee) {
      const int e = ep * 2 + ee;
#pragma unroll
      for (int nt = 0; nt < 4; ++nt) {
        Frag b;
#pragma unroll
        for (int j = 0; j < 8; ++j)
          b.u[j] = lv32[((e * NR + nt * 16 + nB) * KC + kbB + ((j & 4) ? 16 : 0) + (j & 3) * 2) >> 1];
        acc[ee][nt] = __builtin_amdgcn_wmma_f32_16x16x32_bf16(
            false, a.h, false, b.h, (short)0, acc[ee][nt], false, false);
      }
    }
#if !ASYNC_COPY
    __syncthreads();
#endif
  }

  // ---------- per expert: tanh, restage (C/D layout -> A layout), C-GEMM ----
  const unsigned* lds_s32 = (const unsigned*)lds_s[ep];
  const int mbase = rb * 16 + ((lane >> 4) * 8);  // C/D layout: M = mbase + i

#pragma unroll 1
  for (int ee = 0; ee < 2; ++ee) {
    const int e = ep * 2 + ee;
#pragma unroll
    for (int nt = 0; nt < 4; ++nt)
#pragma unroll
      for (int i = 0; i < 8; ++i)
        lds_s[ep][(mbase + i) * NR + nt * 16 + nB] = f2bf(tanhf(acc[ee][nt][i]));
    __syncthreads();   // order LDS stores vs. typed-aliased reads below

    v8f cacc[4];
#pragma unroll
    for (int nt = 0; nt < 4; ++nt) cacc[nt] = vzero();

#pragma unroll
    for (int ks = 0; ks < 2; ++ks) {   // K = R = 64 -> two WMMA steps
      Frag a2;
#pragma unroll
      for (int j = 0; j < 8; ++j) {
        int k = ks * 32 + kbA + ((j & 4) ? 16 : 0) + (j & 3) * 2;
        a2.u[j] = lds_s32[(mrow * NR + k) >> 1];
      }
#pragma unroll
      for (int nt = 0; nt < 4; ++nt) {
        Frag b2;
#pragma unroll
        for (int j = 0; j < 8; ++j) {
          int kk = ks * 32 + kbB + ((j & 4) ? 16 : 0) + (j & 3) * 2;  // B[k][n] = C[e, n, k] (L2-hot, bf16)
          b2.u[j] = *(const unsigned*)(Cbf + (size_t)(e * NR + nt * 16 + nB) * NR + kk);
        }
        cacc[nt] = __builtin_amdgcn_wmma_f32_16x16x32_bf16(
            false, a2.h, false, b2.h, (short)0, cacc[nt], false, false);
      }
    }
#pragma unroll
    for (int nt = 0; nt < 4; ++nt)
#pragma unroll
      for (int i = 0; i < 8; ++i)
        c_out[((size_t)(row0 + mbase + i) * NE + e) * NR + nt * 16 + nB] =
            f2bf(tanhf(cacc[nt][i]));
    __syncthreads();   // WAR: before next expert overwrites lds_s
  }
}

// ============================================================================
// Kernel 2: u_e = U_e · c_e ; out = act( (Σ_e g_e u_e + bias) * x0 + xl ),
// also emits next layer's xl in bf16.  Tile: 128 rows x 64 D-cols per WG.
// ============================================================================
__global__ __launch_bounds__(256)
__attribute__((amdgpu_waves_per_eu(1)))
void crossnet_combine_kernel(
    const bf16_t* __restrict__ c_in,    // [B,E,R] bf16
    const float*  __restrict__ g_in,    // [B,E]
    const bf16_t* __restrict__ Ubf,     // [E,D,R] bf16
    const float*  __restrict__ bias,    // [D]
    const float*  __restrict__ x0,      // [B,D]
    const float*  __restrict__ xl,      // [B,D]
    float*  __restrict__ xout,          // [B,D] f32
    bf16_t* __restrict__ xbf_out,       // [B,D] bf16 mirror (null on last layer)
    int apply_tanh)
{
  const int tid  = threadIdx.x;
  const int lane = tid & 31;
  const int wave = tid >> 5;
  const int row0 = blockIdx.x * TB2;
  const int d0   = blockIdx.y * 64;

  const int mrow  = row0 + wave * 16 + (lane & 15);         // A-layout row
  const int kbA   = (lane >> 4) * 8;
  const int nB    = lane & 15;
  const int kbB   = (lane >> 4) * 8;
  const int mbase = row0 + wave * 16 + ((lane >> 4) * 8);   // C/D-layout row base

  v8f acc[4];
#pragma unroll
  for (int nt = 0; nt < 4; ++nt) acc[nt] = vzero();

#pragma unroll 1
  for (int e = 0; e < NE; ++e) {
    Frag a0, a1;
    const bf16_t* cp = c_in + ((size_t)mrow * NE + e) * NR;
#pragma unroll
    for (int j = 0; j < 8; ++j) {
      int k = kbA + ((j & 4) ? 16 : 0) + (j & 3) * 2;
      a0.u[j] = *(const unsigned*)(cp + k);
      a1.u[j] = *(const unsigned*)(cp + k + 32);
    }
    float ge[8];
#pragma unroll
    for (int i = 0; i < 8; ++i) ge[i] = g_in[(size_t)(mbase + i) * NE + e];

    v8f uacc[4];
#pragma unroll
    for (int nt = 0; nt < 4; ++nt) uacc[nt] = vzero();
#pragma unroll
    for (int nt = 0; nt < 4; ++nt) {
#pragma unroll
      for (int ks = 0; ks < 2; ++ks) {
        Frag b;
#pragma unroll
        for (int j = 0; j < 8; ++j) {
          int kk = ks * 32 + kbB + ((j & 4) ? 16 : 0) + (j & 3) * 2;  // B[k=r][n=d] = U[e, d, r] (bf16)
          b.u[j] = *(const unsigned*)(
              Ubf + (size_t)(e * ND + d0 + nt * 16 + nB) * NR + kk);
        }
        uacc[nt] = __builtin_amdgcn_wmma_f32_16x16x32_bf16(
            false, (ks ? a1.h : a0.h), false, b.h, (short)0, uacc[nt], false, false);
      }
#pragma unroll
      for (int i = 0; i < 8; ++i) acc[nt][i] += ge[i] * uacc[nt][i];
    }
  }

  // epilogue: bias + residual + optional tanh; dual f32/bf16 store
#pragma unroll
  for (int nt = 0; nt < 4; ++nt) {
    int n = d0 + nt * 16 + nB;
    float bv = bias[n];
#pragma unroll
    for (int i = 0; i < 8; ++i) {
      size_t idx = (size_t)(mbase + i) * ND + n;
      float val = (acc[nt][i] + bv) * x0[idx] + xl[idx];
      if (apply_tanh) val = tanhf(val);
      xout[idx] = val;
      if (xbf_out) xbf_out[idx] = f2bf(val);
    }
  }
}

// ============================================================================
extern "C" void kernel_launch(void* const* d_in, const int* in_sizes, int n_in,
                              void* d_out, int out_size, void* d_ws, size_t ws_size,
                              hipStream_t stream) {
  const float* x  = (const float*)d_in[0];   // [B,D]
  const float* U  = (const float*)d_in[1];   // [L,E,D,R]
  const float* V  = (const float*)d_in[2];   // [L,E,R,D]
  const float* Cw = (const float*)d_in[3];   // [L,E,R,R]
  const float* Wg = (const float*)d_in[4];   // [L,E,D]
  const float* bs = (const float*)d_in[5];   // [L,D]
  float* out = (float*)d_out;

  char* p = (char*)d_ws;
  float*  xlbuf = (float*)p;        p += (size_t)NB * ND * 4;      // 64 MB
  bf16_t* xbf   = (bf16_t*)p;       p += (size_t)NB * ND * 2;      // 32 MB
  bf16_t* cbuf  = (bf16_t*)p;       p += (size_t)NB * NE * NR * 2; // 8 MB
  float*  gbuf  = (float*)p;        p += (size_t)NB * NE * 4;      // 256 KB
  bf16_t* vbf   = (bf16_t*)p;       p += (size_t)NL * NE * NR * ND * 2;
  bf16_t* ubf   = (bf16_t*)p;       p += (size_t)NL * NE * ND * NR * 2;
  bf16_t* cwbf  = (bf16_t*)p;       p += (size_t)NL * NE * NR * NR * 2;

  dim3 blk(256);
  // prologue conversions (once per launch)
  cvt_f32_bf16_kernel<<<(NB * ND) / 1024, blk, 0, stream>>>(x, xbf, NB * ND);
  cvt_f32_bf16_kernel<<<(NL * NE * NR * ND) / 1024, blk, 0, stream>>>(
      V, vbf, NL * NE * NR * ND);
  cvt_f32_bf16_kernel<<<(NL * NE * ND * NR) / 1024, blk, 0, stream>>>(
      U, ubf, NL * NE * ND * NR);
  cvt_f32_bf16_kernel<<<(NL * NE * NR * NR) / 1024, blk, 0, stream>>>(
      Cw, cwbf, NL * NE * NR * NR);

  dim3 g1(NB / TB1);
  dim3 g2(NB / TB2, ND / 64);
  for (int l = 0; l < NL; ++l) {
    const float* xin = (l == 0) ? x : xlbuf;
    float* xo = (l == NL - 1) ? out : xlbuf;
    bf16_t* xbf_next = (l == NL - 1) ? (bf16_t*)nullptr : xbf;
    crossnet_vcg_kernel<<<g1, blk, 0, stream>>>(
        xin, xbf, vbf + (size_t)l * NE * NR * ND, cwbf + (size_t)l * NE * NR * NR,
        Wg + (size_t)l * NE * ND, cbuf, gbuf);
    crossnet_combine_kernel<<<g2, blk, 0, stream>>>(
        cbuf, gbuf, ubf + (size_t)l * NE * ND * NR, bs + (size_t)l * ND,
        x, xin, xo, xbf_next, (l < NL - 1) ? 1 : 0);
  }
  (void)in_sizes; (void)n_in; (void)out_size; (void)ws_size;
}
